// Attention_13829794693777
// MI455X (gfx1250) — matrix operand order repaired and checked by reading, not compiled, not run on hardware
//
#include <hip/hip_runtime.h>
#include <hip/hip_bf16.h>

typedef __attribute__((ext_vector_type(16))) _Float16 v16h;
typedef __attribute__((ext_vector_type(8)))  float    v8f;
typedef __attribute__((ext_vector_type(4)))  int      v4i;

// fragment <-> memory bridge: v16h = two disjoint 16-byte chunks
union Frag16 { v16h v; uint4 u[2]; };
union H4     { _Float16 h[4]; uint2 u; };

#define DIM      384
#define NHEADS   8
#define HD       48
#define HDP      64     // Q/K head dim padded to 64 (zeros in 48..63)
#define NTOK     4096   // tokens per batch (64x64)
#define BATCH    2
#define M_TOTAL  (BATCH*NTOK)   // 8192
#define QKV_OUT  (3*DIM)        // 1152

// ---------------- workspace layout (bytes) ----------------
#define OFF_X16  0                               // 8192*384*2      = 6,291,456
#define OFF_W16  6291456                         // 1152*384*2      =   884,736
#define OFF_Q16  7176192                         // 2*8*4096*64*2   = 8,388,608
#define OFF_K16  15564800                        // 2*8*4096*64*2   = 8,388,608
#define OFF_VT   23953408                        // 2*8*48*4096*2   = 6,291,456 (d-major V)
#define OFF_VF   30244864                        // 2*4096*384*4    =12,582,912
// total ~42.8 MB

// ---------------- async global->LDS copy (CDNA5 path, guarded) -------------
#if defined(__gfx1250__) && __has_builtin(__builtin_amdgcn_global_load_async_to_lds_b128)
#define HAVE_ASYNC_LDS 1
#define ASYNC_CP16(gsrc, ldst)                                                  \
  __builtin_amdgcn_global_load_async_to_lds_b128(                               \
      (v4i*)(gsrc), (v4i*)(ldst), 0, 0)
#if __has_builtin(__builtin_amdgcn_s_wait_asynccnt)
#define WAIT_ASYNC() __builtin_amdgcn_s_wait_asynccnt(0)
#else
#define WAIT_ASYNC() asm volatile("s_wait_asynccnt 0x0" ::: "memory")
#endif
#else
#define HAVE_ASYNC_LDS 0
#define ASYNC_CP16(gsrc, ldst) (*(uint4*)(ldst) = *(const uint4*)(gsrc))
#define WAIT_ASYNC() ((void)0)
#endif

// ============================================================
// Kernel 1: fp32 -> f16 conversions (x, and W kept row-major:
// W[n][k] IS the B-fragment layout, k contiguous)
// ============================================================
__global__ void cvt_inputs_k(const float4* __restrict__ x4,
                             const float4* __restrict__ w4,
                             uint2* __restrict__ x16,
                             uint2* __restrict__ w16) {
  const int i = blockIdx.x * blockDim.x + threadIdx.x;
  if (i < (M_TOTAL * DIM) / 4) {
    float4 f = x4[i]; H4 t;
    t.h[0] = (_Float16)f.x; t.h[1] = (_Float16)f.y;
    t.h[2] = (_Float16)f.z; t.h[3] = (_Float16)f.w;
    x16[i] = t.u;
  }
  if (i < (QKV_OUT * DIM) / 4) {
    float4 f = w4[i]; H4 t;
    t.h[0] = (_Float16)f.x; t.h[1] = (_Float16)f.y;
    t.h[2] = (_Float16)f.z; t.h[3] = (_Float16)f.w;
    w16[i] = t.u;
  }
}

// ============================================================
// Kernel 2: zero the padded Q/K f16 region (d=48..63 must be 0)
// ============================================================
__global__ void zero_ws_k(uint4* __restrict__ p, int n16) {
  int i = blockIdx.x * blockDim.x + threadIdx.x;
  if (i < n16) p[i] = make_uint4(0u, 0u, 0u, 0u);
}

// ============================================================
// Kernel 3: QKV projection GEMM  (8192x384) @ W^T(384x1152) + bias
//   block tile 128x64. B column-block (64 cols x full K=384) staged
//   in LDS once (async); A fragments straight from global (L2-resident).
// ============================================================
#define GT_M 128
#define GT_N 64

__global__ __launch_bounds__(256) void qkv_gemm_k(
    const _Float16* __restrict__ A,   // x16 (8192,384) row-major
    const _Float16* __restrict__ W,   // w16 (1152,384) row-major = frag layout
    const float*    __restrict__ bias,
    float scale,
    _Float16* __restrict__ Qh, _Float16* __restrict__ Kh,
    _Float16* __restrict__ Vt,        // d-major V: [b*8+h][d<48][n]
    float* __restrict__ Vf)
{
  __shared__ _Float16 Bs[GT_N][DIM];   // 48 KB: 64 output cols x K=384

  const int tid  = threadIdx.x;
  const int wave = tid >> 5, lane = tid & 31;
  const int m0   = blockIdx.x * GT_M;
  const int n0   = blockIdx.y * GT_N;

  // stage the 64-column weight block once: 64*384 halves = 3072 x b128
  #pragma unroll
  for (int t = tid; t < 3072; t += 256) {
    const int r = t / 48, c = (t % 48) * 8;
    ASYNC_CP16(&W[(size_t)(n0 + r) * DIM + c], &Bs[r][c]);
  }
  WAIT_ASYNC();
  __syncthreads();

  v8f acc[4] = {};
  const int mrow = m0 + wave * 16 + (lane & 15);
  const int akb  = (lane < 16) ? 0 : 16;  // A-layout K base
  const int bn   = lane & 15;
  const int bkb  = (lane < 16) ? 0 : 16;  // B-layout K base
  const _Float16* arow = A + (size_t)mrow * DIM;

  for (int k0 = 0; k0 < DIM; k0 += 32) {
    if (k0 + 32 < DIM)
      __builtin_prefetch(&arow[k0 + 32], 0, 1);   // global_prefetch_b8
    Frag16 af;   // A fragment: K chunk {kb..kb+15}
    af.u[0] = *(const uint4*)&arow[k0 + akb];
    af.u[1] = *(const uint4*)&arow[k0 + akb + 8];
    #pragma unroll
    for (int t = 0; t < 4; ++t) {
      const v16h bf = *(const v16h*)&Bs[t * 16 + bn][k0 + bkb];  // contiguous, 32B aligned
      acc[t] = __builtin_amdgcn_wmma_f32_16x16x32_f16(false, af.v, false, bf,
                                                      (short)0, acc[t], false, false);
    }
  }

  // epilogue: bias add + scatter into Q*scale / K (padded f16), V (d-major f16 + f32)
  const int colb = lane & 15, rowh = (lane >> 4) * 8;
  #pragma unroll
  for (int t = 0; t < 4; ++t) {
    const int o   = n0 + t * 16 + colb;          // output column 0..1151
    const float bv = bias[o];
    const int s = o / DIM, rem = o % DIM, h = rem / HD, d = rem % HD;
    #pragma unroll
    for (int i = 0; i < 8; ++i) {
      const int m = m0 + wave * 16 + rowh + i;   // flat token 0..8191
      const float val = acc[t][i] + bv;
      const int b = m >> 12, n = m & (NTOK - 1);
      const int bh = b * NHEADS + h;
      const size_t qi = ((size_t)(bh * NTOK + n)) * HDP + d;
      if (s == 0)      Qh[qi] = (_Float16)(val * scale);
      else if (s == 1) Kh[qi] = (_Float16)val;
      else {
        Vt[((size_t)bh * HD + d) * NTOK + n] = (_Float16)val;   // d-major
        Vf[(size_t)m * DIM + rem] = val;
      }
    }
  }
}

// ============================================================
// Kernel 4: flash attention, one (b,h) per blockIdx.y,
//   8 waves x 16 queries = 128 queries/block, key blocks of 32,
//   double-buffered async K/V staging, 1 barrier per iteration
// ============================================================
__global__ __launch_bounds__(256) void flash_attn_k(
    const _Float16* __restrict__ Qh, const _Float16* __restrict__ Kh,
    const _Float16* __restrict__ VT, float* __restrict__ Out)
{
  __shared__ _Float16 Ks[2][32][HDP];   // 8 KB  [buf][key][d]
  __shared__ _Float16 Vs[2][HD][32];    // 6 KB  [buf][d][key]  (d-major from global)
  __shared__ _Float16 Ps[8][16][32];    // 8 KB  per-wave P staging

  const int bh  = blockIdx.y;                 // b*8 + h
  const int tid = threadIdx.x, wave = tid >> 5, lane = tid & 31;
  const _Float16* Qb = Qh + (size_t)bh * NTOK * HDP;
  const _Float16* Kb = Kh + (size_t)bh * NTOK * HDP;
  const _Float16* Vb = VT + (size_t)bh * HD * NTOK;
  const int q0 = blockIdx.x * 128 + wave * 16;

  // cooperative tile staging: K tile 32x64 (256 x b128), V tile 48x32 (192 x b128)
  const int kr = tid >> 3, kc = (tid & 7) * 8;   // K: row, col8
  const int vd = tid >> 2, vc = (tid & 3) * 8;   // V: d, key8
  auto stage = [&](int buf, int j) {
    ASYNC_CP16(&Kb[(size_t)(j + kr) * HDP + kc], &Ks[buf][kr][kc]);
    if (tid < 192)
      ASYNC_CP16(&Vb[(size_t)vd * NTOK + j + vc], &Vs[buf][vd][vc]);
  };

  // Q fragments (A-layout), K-chunks 0..31 and 32..63 — vector loads
  Frag16 aq0, aq1;
  {
    const int m = lane & 15, kb = (lane < 16) ? 0 : 16;
    const _Float16* qr = Qb + (size_t)(q0 + m) * HDP;
    aq0.u[0] = *(const uint4*)&qr[kb];
    aq0.u[1] = *(const uint4*)&qr[kb + 8];
    aq1.u[0] = *(const uint4*)&qr[32 + kb];
    aq1.u[1] = *(const uint4*)&qr[40 + kb];
  }

  v8f o[3] = {};                 // out accum, d tiles 0..2 (d<48)
  float mrow[8], lrow[8];
  #pragma unroll
  for (int i = 0; i < 8; ++i) { mrow[i] = -1e30f; lrow[i] = 0.f; }

  stage(0, 0);
  int cur = 0;
  for (int j = 0; j < NTOK; j += 32) {
    WAIT_ASYNC();          // my async writes into Ks/Vs[cur] done
    __syncthreads();       // everyone's staging done; prev buffer free
    const int nxt = cur ^ 1;
    if (j + 32 < NTOK) stage(nxt, j + 32);   // stream next tile during compute

    // ---- S = Q * K^T  (two 16x16 tiles, K-chain over padded 64) ----
    v8f s0 = {}, s1 = {};
    {
      const int col = lane & 15, kb = (lane < 16) ? 0 : 16;
      const v16h bk00 = *(const v16h*)&Ks[cur][col][kb];
      const v16h bk01 = *(const v16h*)&Ks[cur][col][32 + kb];
      const v16h bk10 = *(const v16h*)&Ks[cur][col + 16][kb];
      const v16h bk11 = *(const v16h*)&Ks[cur][col + 16][32 + kb];
      s0 = __builtin_amdgcn_wmma_f32_16x16x32_f16(false, aq0.v, false, bk00, (short)0, s0, false, false);
      s0 = __builtin_amdgcn_wmma_f32_16x16x32_f16(false, aq1.v, false, bk01, (short)0, s0, false, false);
      s1 = __builtin_amdgcn_wmma_f32_16x16x32_f16(false, aq0.v, false, bk10, (short)0, s1, false, false);
      s1 = __builtin_amdgcn_wmma_f32_16x16x32_f16(false, aq1.v, false, bk11, (short)0, s1, false, false);
    }

    // ---- online softmax (row = VGPR index, cols spread over 16-lane halves) ----
    float p0[8], p1[8];
    #pragma unroll
    for (int i = 0; i < 8; ++i) {
      float t = fmaxf(s0[i], s1[i]);
      #pragma unroll
      for (int off = 1; off < 16; off <<= 1) t = fmaxf(t, __shfl_xor(t, off));
      const float mnew  = fmaxf(mrow[i], t);
      const float alpha = __expf(mrow[i] - mnew);
      mrow[i] = mnew;
      p0[i] = __expf(s0[i] - mnew);
      p1[i] = __expf(s1[i] - mnew);
      float rs = p0[i] + p1[i];
      #pragma unroll
      for (int off = 1; off < 16; off <<= 1) rs += __shfl_xor(rs, off);
      lrow[i] = lrow[i] * alpha + rs;
      o[0][i] *= alpha; o[1][i] *= alpha; o[2][i] *= alpha;
    }

    // ---- re-fragment P: C-layout -> LDS -> A-layout (per-wave patch) ----
    {
      const int col = lane & 15, rh = (lane >> 4) * 8;
      #pragma unroll
      for (int i = 0; i < 8; ++i) {
        Ps[wave][rh + i][col]      = (_Float16)p0[i];
        Ps[wave][rh + i][16 + col] = (_Float16)p1[i];
      }
    }
    Frag16 ap;
    {
      const int m = lane & 15, kb = (lane < 16) ? 0 : 16;
      ap.u[0] = *(const uint4*)&Ps[wave][m][kb];
      ap.u[1] = *(const uint4*)&Ps[wave][m][kb + 8];
    }

    // ---- O += P * V (3 d-tiles, V is d-major so B-frags are contiguous) ----
    {
      const int col = lane & 15, kb = (lane < 16) ? 0 : 16;
      #pragma unroll
      for (int t = 0; t < 3; ++t) {
        const v16h bv = *(const v16h*)&Vs[cur][t * 16 + col][kb];
        o[t] = __builtin_amdgcn_wmma_f32_16x16x32_f16(false, ap.v, false, bv, (short)0, o[t], false, false);
      }
    }
    cur = nxt;
  }

  // ---- normalize + write out ----
  {
    const int col = lane & 15, rh = (lane >> 4) * 8;
    const int b = bh >> 3, h = bh & 7;
    #pragma unroll
    for (int t = 0; t < 3; ++t) {
      const int d = t * 16 + col;
      #pragma unroll
      for (int i = 0; i < 8; ++i) {
        const int m = q0 + rh + i;
        Out[((size_t)(b * NTOK + m)) * DIM + h * HD + d] = o[t][i] / lrow[i];
      }
    }
  }
}

// ============================================================
// Kernel 5: LePE depthwise 3x3 conv + bias, 4 channels/thread,
//   accumulate into Out
// ============================================================
__global__ void lepe_add_k(const float* __restrict__ Vf,
                           const float* __restrict__ w,
                           const float* __restrict__ b,
                           float* __restrict__ Out) {
  const int idx = blockIdx.x * blockDim.x + threadIdx.x;   // over (B*N*DIM/4)
  if (idx >= BATCH * NTOK * (DIM / 4)) return;
  const int c4 = (idx % (DIM / 4)) * 4;
  const int n  = (idx / (DIM / 4)) & (NTOK - 1);
  const int bb = idx / ((DIM / 4) * NTOK);
  const int y = n >> 6, x = n & 63;

  float4 acc = *(const float4*)&b[c4];
  #pragma unroll
  for (int dy = -1; dy <= 1; ++dy) {
    const int yy = y + dy;
    if (yy < 0 || yy >= 64) continue;
    #pragma unroll
    for (int dx = -1; dx <= 1; ++dx) {
      const int xx = x + dx;
      if (xx < 0 || xx >= 64) continue;
      const float4 vv = *(const float4*)&Vf[((size_t)(bb * NTOK + (yy << 6) + xx)) * DIM + c4];
      const int wi = (dy + 1) * 3 + (dx + 1);
      acc.x += w[(c4 + 0) * 9 + wi] * vv.x;
      acc.y += w[(c4 + 1) * 9 + wi] * vv.y;
      acc.z += w[(c4 + 2) * 9 + wi] * vv.z;
      acc.w += w[(c4 + 3) * 9 + wi] * vv.w;
    }
  }
  float4* op = (float4*)&Out[((size_t)(bb * NTOK + n)) * DIM + c4];
  float4 cur = *op;
  cur.x += acc.x; cur.y += acc.y; cur.z += acc.z; cur.w += acc.w;
  *op = cur;
}

// ============================================================
extern "C" void kernel_launch(void* const* d_in, const int* in_sizes, int n_in,
                              void* d_out, int out_size, void* d_ws, size_t ws_size,
                              hipStream_t stream) {
  const float* x      = (const float*)d_in[0];
  const float* qkv_w  = (const float*)d_in[1];
  const float* qkv_b  = (const float*)d_in[2];
  const float* lepe_w = (const float*)d_in[3];
  const float* lepe_b = (const float*)d_in[4];
  float* out = (float*)d_out;

  char* ws = (char*)d_ws;
  _Float16* x16 = (_Float16*)(ws + OFF_X16);
  _Float16* w16 = (_Float16*)(ws + OFF_W16);
  _Float16* Qh  = (_Float16*)(ws + OFF_Q16);
  _Float16* Kh  = (_Float16*)(ws + OFF_K16);
  _Float16* Vt  = (_Float16*)(ws + OFF_VT);
  float*    Vf  = (float*)   (ws + OFF_VF);

  const float scale = 0.14433756729740643f;   // 48^-0.5

  // 1) convert inputs to f16 (x; W stays row-major = B-fragment layout)
  cvt_inputs_k<<<((M_TOTAL * DIM / 4) + 255) / 256, 256, 0, stream>>>(
      (const float4*)x, (const float4*)qkv_w, (uint2*)x16, (uint2*)w16);

  // 2) zero padded Q/K region (2 * 8,388,608 bytes)
  {
    const int n16 = (2 * 8388608) / 16;
    zero_ws_k<<<(n16 + 255) / 256, 256, 0, stream>>>((uint4*)(ws + OFF_Q16), n16);
  }

  // 3) QKV GEMM
  {
    dim3 grid(M_TOTAL / GT_M, QKV_OUT / GT_N);   // 64 x 18
    qkv_gemm_k<<<grid, 256, 0, stream>>>(x16, w16, qkv_b, scale, Qh, Kh, Vt, Vf);
  }

  // 4) flash attention
  {
    dim3 grid(NTOK / 128, BATCH * NHEADS);       // 32 x 16
    flash_attn_k<<<grid, 256, 0, stream>>>(Qh, Kh, Vt, out);
  }

  // 5) LePE conv + bias accumulate
  lepe_add_k<<<((BATCH * NTOK * DIM / 4) + 255) / 256, 256, 0, stream>>>(
      Vf, lepe_w, lepe_b, out);
}
